// Decoder_60198261621363
// MI455X (gfx1250) — matrix operand order repaired and checked by reading, not compiled, not run on hardware
//
#include <hip/hip_runtime.h>
#include <hip/hip_bf16.h>
#include <math.h>

// ---------------------------------------------------------------------------
// GPT decoder forward for gfx1250 (CDNA5, wave32, WMMA, bf16 operands,
// async global->LDS double-buffered GEMM pipeline).
// Model dims: V=32000 E=1024 L=6 H=16 B=4 T=1024 HS=64 FF=4096
// ---------------------------------------------------------------------------

#define VOCAB 32000
#define EMB   1024
#define NLAYER 6
#define NHEAD 16
#define BATCH 4
#define SEQ   1024
#define HS    64
#define FFD   4096
#define MROWS (BATCH * SEQ)   // 4096

typedef __attribute__((ext_vector_type(16))) __bf16 v16bf;
typedef __attribute__((ext_vector_type(8)))  float  v8f;

union Frag16 {
    unsigned int u[8];
    v16bf v;
};

// float -> bf16 (round-to-nearest-even); no __bf16 arithmetic anywhere.
__device__ __forceinline__ unsigned short f2bf(float f) {
    unsigned int u = __float_as_uint(f);
    u += 0x7FFFu + ((u >> 16) & 1u);
    return (unsigned short)(u >> 16);
}
__device__ __forceinline__ unsigned int pack2bf(float a, float b) {
    return (unsigned int)f2bf(a) | ((unsigned int)f2bf(b) << 16);
}

// 16-bit A-matrix (16x32) slot -> K offset (ISA 7.12.2).
__device__ __forceinline__ int a_slot_k(int s, int hi) {
    return (s < 4 ? 2 * s : 16 + 2 * (s - 4)) + hi * 8;
}
// 16-bit B-matrix (32x16) slot -> K offset.
__device__ __forceinline__ int b_slot_k(int s, int hi) {
    return (s < 4 ? 2 * s : 16 + 2 * (s - 4)) + hi * 8;
}

__device__ __forceinline__ v8f wmma_bf16(const Frag16& a, const Frag16& b, v8f c) {
    return __builtin_amdgcn_wmma_f32_16x16x32_bf16(
        false, a.v, false, b.v, (short)0, c, false, false);
}

// Async 16B global -> LDS copy (per lane).  ASYNCcnt-tracked (ISA 15.18.3 #98).
// LDS address = low 32 bits of the generic pointer (wave-relative LDS offset).
__device__ __forceinline__ void async_b128(const void* g, void* l) {
    asm volatile("global_load_async_to_lds_b128 %0, %1, off"
                 :: "v"((unsigned int)(uintptr_t)l), "v"(g)
                 : "memory");
}
__device__ __forceinline__ void wait_async_le4() {
    asm volatile("s_wait_asynccnt 4" ::: "memory");
}
__device__ __forceinline__ void wait_async_0() {
    asm volatile("s_wait_asynccnt 0" ::: "memory");
}
// wave-local LDS fence (drain DS ops before cross-lane re-read within a wave)
__device__ __forceinline__ void wave_lds_fence() {
    asm volatile("s_wait_dscnt 0" ::: "memory");
}

// butterfly helpers over 16-lane halves via ds_swizzle (xor masks, and=0x1f)
template <int IMM>
__device__ __forceinline__ float swz_f(float x) {
    return __int_as_float(__builtin_amdgcn_ds_swizzle(__float_as_int(x), IMM));
}
__device__ __forceinline__ float red_max16(float x) {
    x = fmaxf(x, swz_f<0x041f>(x));   // xor 1
    x = fmaxf(x, swz_f<0x081f>(x));   // xor 2
    x = fmaxf(x, swz_f<0x101f>(x));   // xor 4
    x = fmaxf(x, swz_f<0x201f>(x));   // xor 8
    return x;
}
__device__ __forceinline__ float red_sum16(float x) {
    x += swz_f<0x041f>(x);
    x += swz_f<0x081f>(x);
    x += swz_f<0x101f>(x);
    x += swz_f<0x201f>(x);
    return x;
}

// ---------------------------------------------------------------------------
// f32 -> bf16 bulk convert (8 elements / thread). n multiple of 2048.
// ---------------------------------------------------------------------------
__global__ void cvt_bf16_kernel(const float* __restrict__ src,
                                unsigned short* __restrict__ dst) {
    const size_t i = ((size_t)blockIdx.x * 256 + threadIdx.x) * 8;
    float4 a = *(const float4*)(src + i);
    float4 b = *(const float4*)(src + i + 4);
    uint4 o;
    o.x = pack2bf(a.x, a.y);
    o.y = pack2bf(a.z, a.w);
    o.z = pack2bf(b.x, b.y);
    o.w = pack2bf(b.z, b.w);
    *(uint4*)(dst + i) = o;
}

// ---------------------------------------------------------------------------
// f32 [K,N] -> bf16 [N,K] transpose-convert.  32x32 tiles via LDS.
// grid = (N/32, K/32), block = 256.
// ---------------------------------------------------------------------------
__global__ void cvt_t_kernel(const float* __restrict__ src,
                             unsigned short* __restrict__ dst,
                             int N, int K) {
    __shared__ unsigned short tile[32 * 33];
    const int tx = threadIdx.x & 31;
    const int ty = threadIdx.x >> 5;          // 0..7
    const int n0 = blockIdx.x * 32;
    const int k0 = blockIdx.y * 32;
#pragma unroll
    for (int i = 0; i < 4; ++i) {
        const int kl = ty + 8 * i;
        tile[kl * 33 + tx] = f2bf(src[(size_t)(k0 + kl) * N + n0 + tx]);
    }
    __syncthreads();
#pragma unroll
    for (int i = 0; i < 4; ++i) {
        const int nl = ty + 8 * i;
        dst[(size_t)(n0 + nl) * K + k0 + tx] = tile[tx * 33 + nl];
    }
}

// ---------------------------------------------------------------------------
// x[b,t,:] = tok_emb[tokens[b,t],:] + pos_emb[t,:]   (f32 residual stream)
// ---------------------------------------------------------------------------
__global__ void embed_kernel(const int* __restrict__ tokens,
                             const float* __restrict__ tok_emb,
                             const float* __restrict__ pos_emb,
                             float* __restrict__ x) {
    const int row = blockIdx.x;
    const int t   = row & (SEQ - 1);
    const int tok = tokens[row];
    const int i   = threadIdx.x * 4;
    float4 a = *(const float4*)(tok_emb + (size_t)tok * EMB + i);
    float4 p = *(const float4*)(pos_emb + (size_t)t * EMB + i);
    a.x += p.x; a.y += p.y; a.z += p.z; a.w += p.w;
    *(float4*)(x + (size_t)row * EMB + i) = a;
}

// ---------------------------------------------------------------------------
// LayerNorm over E=1024, f32 in -> bf16 out. One block (256 threads) per row.
// ---------------------------------------------------------------------------
__global__ void ln_kernel(const float* __restrict__ X,
                          const float* __restrict__ g,
                          const float* __restrict__ b,
                          unsigned short* __restrict__ Y) {
    __shared__ float red[512];
    const int row = blockIdx.x;
    const int tid = threadIdx.x;
    const int i   = tid * 4;
    float4 xv = *(const float4*)(X + (size_t)row * EMB + i);
    red[tid]       = xv.x + xv.y + xv.z + xv.w;
    red[256 + tid] = xv.x * xv.x + xv.y * xv.y + xv.z * xv.z + xv.w * xv.w;
    __syncthreads();
    for (int off = 128; off > 0; off >>= 1) {
        if (tid < off) {
            red[tid] += red[tid + off];
            red[256 + tid] += red[256 + tid + off];
        }
        __syncthreads();
    }
    const float mean = red[0] * (1.0f / EMB);
    const float var  = red[256] * (1.0f / EMB) - mean * mean;
    const float rstd = rsqrtf(var + 1e-5f);
    float4 gv = *(const float4*)(g + i);
    float4 bv = *(const float4*)(b + i);
    uint2 o;
    o.x = pack2bf((xv.x - mean) * rstd * gv.x + bv.x,
                  (xv.y - mean) * rstd * gv.y + bv.y);
    o.y = pack2bf((xv.z - mean) * rstd * gv.z + bv.z,
                  (xv.w - mean) * rstd * gv.w + bv.w);
    *(uint2*)(Y + (size_t)row * EMB + i) = o;
}

// ---------------------------------------------------------------------------
// WMMA GEMM: C[M,N] = f(A[M,K] @ W[K,N] + bias + residual)
// A: bf16 [M,K] row-major; WT: bf16 [N,K] (pre-transposed) -> both operands are
// staged with pure async global->LDS B128 copies, double-buffered so tile kt+1
// streams in while tile kt is consumed by WMMA.
// Block tile 128x128, BK=32, 8 waves (2Mx4N), 8 v_wmma per K-step per wave.
// FLAGS: bit0 +bias, bit1 relu, bit2 +residual(f32), bit3 bf16 output.
// ---------------------------------------------------------------------------
#define GF_BIAS  1
#define GF_RELU  2
#define GF_RES   4
#define GF_OUTBF 8

#define LDS_STRIDE 40            // padded bf16 row stride (80B, 16B-aligned)
#define TILE_USHORTS (128 * LDS_STRIDE)

template <int FLAGS>
__global__ void gemm_kernel(const unsigned short* __restrict__ A,
                            const unsigned short* __restrict__ WT,
                            const float* __restrict__ bias,
                            const float* __restrict__ residual,
                            void* __restrict__ Cout,
                            int M, int N, int K) {
    __shared__ unsigned short As[2 * TILE_USHORTS];  // [buf][row][k]
    __shared__ unsigned short Bs[2 * TILE_USHORTS];  // [buf][col][k]

    const int tid  = threadIdx.x;
    const int lane = tid & 31;
    const int wid  = tid >> 5;
    const int wm   = wid & 1;      // 64-row band
    const int wn   = wid >> 1;     // 32-col band (0..3)
    const int lr   = lane & 15;
    const int hi   = lane >> 4;

    const int n0 = blockIdx.x * 128;
    const int m0 = blockIdx.y * 128;

    // staging: thread -> (row 0..127, 2 x 16B chunks covering the 64B row)
    const int st_row = tid >> 1;
    const int st_c0  = (tid & 1) * 16;      // element offset {0,16}

    const unsigned short* gA = A  + (size_t)(m0 + st_row) * K + st_c0;
    const unsigned short* gB = WT + (size_t)(n0 + st_row) * K + st_c0;
    unsigned short* lA = &As[st_row * LDS_STRIDE + st_c0];
    unsigned short* lB = &Bs[st_row * LDS_STRIDE + st_c0];

    auto issue_tile = [&](int kt, int buf) {
        const unsigned short* ga = gA + kt * 32;
        const unsigned short* gb = gB + kt * 32;
        unsigned short* la = lA + buf * TILE_USHORTS;
        unsigned short* lb = lB + buf * TILE_USHORTS;
        async_b128(ga,     la);
        async_b128(ga + 8, la + 8);
        async_b128(gb,     lb);
        async_b128(gb + 8, lb + 8);
    };

    v8f acc[4][2];
#pragma unroll
    for (int i = 0; i < 4; ++i) { acc[i][0] = v8f{}; acc[i][1] = v8f{}; }

    const int ksteps = K >> 5;
    issue_tile(0, 0);

    for (int kt = 0; kt < ksteps; ++kt) {
        const int cur = kt & 1;
        const bool more = (kt + 1 < ksteps);
        if (more) {
            issue_tile(kt + 1, cur ^ 1);  // stream next tile while computing
            wait_async_le4();             // own copies of tile kt complete
        } else {
            wait_async_0();
        }
        __syncthreads();                  // all waves' copies complete

        const unsigned short* asb = &As[cur * TILE_USHORTS];
        const unsigned short* bsb = &Bs[cur * TILE_USHORTS];

        Frag16 af[4], bf[2];
#pragma unroll
        for (int i = 0; i < 4; ++i) {
            const int row = wm * 64 + i * 16 + lr;
#pragma unroll
            for (int s = 0; s < 8; ++s)
                af[i].u[s] =
                    *(const unsigned int*)&asb[row * LDS_STRIDE + a_slot_k(s, hi)];
        }
#pragma unroll
        for (int j = 0; j < 2; ++j) {
            const int col = wn * 32 + j * 16 + lr;
#pragma unroll
            for (int s = 0; s < 8; ++s)
                bf[j].u[s] =
                    *(const unsigned int*)&bsb[col * LDS_STRIDE + b_slot_k(s, hi)];
        }
#pragma unroll
        for (int i = 0; i < 4; ++i)
#pragma unroll
            for (int j = 0; j < 2; ++j)
                acc[i][j] = wmma_bf16(af[i], bf[j], acc[i][j]);

        __syncthreads();                  // readers done before buffer reuse
    }

    // ---- epilogue (straight-line per instantiation) ----
#pragma unroll
    for (int i = 0; i < 4; ++i) {
#pragma unroll
        for (int j = 0; j < 2; ++j) {
            const int col  = n0 + wn * 32 + j * 16 + lr;
            const int rowb = m0 + wm * 64 + i * 16 + hi * 8;
            float bv = 0.0f;
            if constexpr (FLAGS & GF_BIAS) bv = bias[col];
#pragma unroll
            for (int r = 0; r < 8; ++r) {
                const size_t idx = (size_t)(rowb + r) * N + col;
                float val = acc[i][j][r] + bv;
                if constexpr (FLAGS & GF_RES)  val += residual[idx];
                if constexpr (FLAGS & GF_RELU) val = fmaxf(val, 0.0f);
                if constexpr (FLAGS & GF_OUTBF)
                    ((unsigned short*)Cout)[idx] = f2bf(val);
                else
                    ((float*)Cout)[idx] = val;
            }
        }
    }
}

// ---------------------------------------------------------------------------
// Causal flash attention, bf16 q/k/v in, bf16 out.
// 128-thread blocks = 4 independent waves; wave w owns q rows
// [(blockIdx.x*4+w)*16, +16).  No workgroup barriers (divergent causal trip
// counts) -- cross-lane P re-layout uses a wave-local DS fence.
// scale = E^-0.5 (reference scales by emb dim, not head dim).
// ---------------------------------------------------------------------------
__global__ void attn_kernel(const unsigned short* __restrict__ Q,
                            const unsigned short* __restrict__ Km,
                            const unsigned short* __restrict__ Vm,
                            unsigned short* __restrict__ O,
                            float scale) {
    __shared__ unsigned short Ps[4][16 * 40];  // per-wave 16x32 bf16 P tile

    const int tid   = threadIdx.x;
    const int lane  = tid & 31;
    const int wid   = tid >> 5;
    const int lr    = lane & 15;
    const int hi    = lane >> 4;
    const int h     = blockIdx.y;
    const int b     = blockIdx.z;
    const int qbase = (blockIdx.x * 4 + wid) * 16;

    unsigned short* ps = &Ps[wid][0];

    // Q fragments (A operand): adjacent bf16 pairs -> direct u32 loads
    Frag16 aq[2];
    {
        const unsigned short* qp =
            Q + ((size_t)(b * SEQ + qbase + lr)) * EMB + h * HS;
#pragma unroll
        for (int kc = 0; kc < 2; ++kc)
#pragma unroll
            for (int s = 0; s < 8; ++s)
                aq[kc].u[s] = *(const unsigned int*)(qp + kc * 32 + a_slot_k(s, hi));
    }

    float mrow[8], lrow[8];
    v8f o[4];
#pragma unroll
    for (int r = 0; r < 8; ++r) { mrow[r] = -1e30f; lrow[r] = 0.0f; }
    o[0] = v8f{}; o[1] = v8f{}; o[2] = v8f{}; o[3] = v8f{};

    const int nblk = (qbase + 47) >> 5;
    for (int jb = 0; jb < nblk; ++jb) {
        const int j0 = jb * 32;

        if (jb + 1 < nblk) {   // prefetch next KV block (global_prefetch_b8)
            __builtin_prefetch(
                Km + ((size_t)(b * SEQ + j0 + 32 + lane)) * EMB + h * HS, 0, 3);
            __builtin_prefetch(
                Vm + ((size_t)(b * SEQ + j0 + 32 + lane)) * EMB + h * HS, 0, 3);
        }

        // S = Q @ K^T, one 16x16 column tile at a time (K-frags built JIT to
        // keep register pressure low -- avoids scratch spills)
        v8f sc[2];
#pragma unroll
        for (int n = 0; n < 2; ++n) {
            const unsigned short* kp =
                Km + ((size_t)(b * SEQ + j0 + 16 * n + lr)) * EMB + h * HS;
            Frag16 bk;
#pragma unroll
            for (int s = 0; s < 8; ++s)
                bk.u[s] = *(const unsigned int*)(kp + b_slot_k(s, hi));
            sc[n] = wmma_bf16(aq[0], bk, v8f{});
#pragma unroll
            for (int s = 0; s < 8; ++s)
                bk.u[s] = *(const unsigned int*)(kp + 32 + b_slot_k(s, hi));
            sc[n] = wmma_bf16(aq[1], bk, sc[n]);
        }

        const int col0 = j0 + lr;
        const int col1 = j0 + 16 + lr;
#pragma unroll
        for (int r = 0; r < 8; ++r) {
            const int row = qbase + r + 8 * hi;
            float x0 = sc[0][r] * scale; if (col0 > row) x0 = -1e30f;
            float x1 = sc[1][r] * scale; if (col1 > row) x1 = -1e30f;
            const float mx = red_max16(fmaxf(x0, x1));
            const float mn = fmaxf(mrow[r], mx);
            const float p0 = __expf(x0 - mn);
            const float p1 = __expf(x1 - mn);
            const float pssum = red_sum16(p0 + p1);
            const float alpha = __expf(mrow[r] - mn);
            lrow[r] = lrow[r] * alpha + pssum;
            mrow[r] = mn;
            o[0][r] *= alpha; o[1][r] *= alpha; o[2][r] *= alpha; o[3][r] *= alpha;
            const int prow = r + 8 * hi;
            ps[prow * 40 + lr]      = f2bf(p0);
            ps[prow * 40 + 16 + lr] = f2bf(p1);
        }
        wave_lds_fence();

        // P as A-fragment (16 rows x 32 kv)
        Frag16 ap;
#pragma unroll
        for (int s = 0; s < 8; ++s)
            ap.u[s] = *(const unsigned int*)&ps[lr * 40 + a_slot_k(s, hi)];

        // V fragments (B operand): col = d, slot = kv (stride-E pair gather)
#pragma unroll
        for (int j = 0; j < 4; ++j) {
            Frag16 bv;
            const int d = h * HS + 16 * j + lr;
#pragma unroll
            for (int s = 0; s < 8; ++s) {
                const unsigned short* vp =
                    Vm + ((size_t)(b * SEQ + j0 + b_slot_k(s, hi))) * EMB + d;
                bv.u[s] = (unsigned int)vp[0] | ((unsigned int)vp[EMB] << 16);
            }
            o[j] = wmma_bf16(ap, bv, o[j]);
        }
        wave_lds_fence();
    }

    // normalize + write bf16 out
#pragma unroll
    for (int r = 0; r < 8; ++r) lrow[r] = 1.0f / lrow[r];
#pragma unroll
    for (int j = 0; j < 4; ++j) {
        const int d = h * HS + 16 * j + lr;
#pragma unroll
        for (int r = 0; r < 8; ++r) {
            const int row = qbase + r + 8 * hi;
            O[((size_t)(b * SEQ + row)) * EMB + d] = f2bf(o[j][r] * lrow[r]);
        }
    }
}

// ---------------------------------------------------------------------------
// Host-side orchestration
// ---------------------------------------------------------------------------
static inline void cvt(const float* src, unsigned short* dst, size_t n,
                       hipStream_t stream) {
    cvt_bf16_kernel<<<(unsigned)(n / 2048), 256, 0, stream>>>(src, dst);
}
static inline void cvtT(const float* src, unsigned short* dst, int N, int K,
                        hipStream_t stream) {
    cvt_t_kernel<<<dim3(N / 32, K / 32), 256, 0, stream>>>(src, dst, N, K);
}

extern "C" void kernel_launch(void* const* d_in, const int* in_sizes, int n_in,
                              void* d_out, int out_size, void* d_ws, size_t ws_size,
                              hipStream_t stream) {
    const int*   tokens  = (const int*)  d_in[0];
    const float* tok_emb = (const float*)d_in[1];
    const float* pos_emb = (const float*)d_in[2];
    const float* Wq      = (const float*)d_in[3];
    const float* Wk      = (const float*)d_in[4];
    const float* Wv      = (const float*)d_in[5];
    const float* Wo      = (const float*)d_in[6];
    const float* ln1_g   = (const float*)d_in[7];
    const float* ln1_b   = (const float*)d_in[8];
    const float* W1      = (const float*)d_in[9];
    const float* b1      = (const float*)d_in[10];
    const float* W2      = (const float*)d_in[11];
    const float* b2      = (const float*)d_in[12];
    const float* ln2_g   = (const float*)d_in[13];
    const float* ln2_b   = (const float*)d_in[14];
    const float* lmh_w   = (const float*)d_in[15];
    const float* lmh_b   = (const float*)d_in[16];
    float* out = (float*)d_out;

    // workspace: x (f32) | bf16: h,q,k,v,attn (B*T*E), ff (B*T*FF),
    //            wbuf (rotating transposed bf16 weight scratch, E*V elems)
    const size_t NTE = (size_t)MROWS * EMB;
    float*          x    = (float*)d_ws;
    unsigned short* h    = (unsigned short*)(x + NTE);
    unsigned short* q    = h + NTE;
    unsigned short* k    = q + NTE;
    unsigned short* v    = k + NTE;
    unsigned short* attn = v + NTE;
    unsigned short* ff   = attn + NTE;
    unsigned short* wbuf = ff + (size_t)MROWS * FFD;

    const dim3 gemmE (EMB  / 128, MROWS / 128);
    const dim3 gemmFF(FFD  / 128, MROWS / 128);
    const dim3 gemmV (VOCAB/ 128, MROWS / 128);
    const float scale = 0.03125f;   // E^-0.5

    embed_kernel<<<MROWS, 256, 0, stream>>>(tokens, tok_emb, pos_emb, x);

    for (int l = 0; l < NLAYER; ++l) {
        const size_t wEE = (size_t)l * EMB * EMB;
        ln_kernel<<<MROWS, 256, 0, stream>>>(x, ln1_g + l * EMB, ln1_b + l * EMB, h);

        cvtT(Wq + wEE, wbuf, EMB, EMB, stream);
        gemm_kernel<GF_OUTBF><<<gemmE, 256, 0, stream>>>(h, wbuf, nullptr, nullptr, q,
                                                         MROWS, EMB, EMB);
        cvtT(Wk + wEE, wbuf, EMB, EMB, stream);
        gemm_kernel<GF_OUTBF><<<gemmE, 256, 0, stream>>>(h, wbuf, nullptr, nullptr, k,
                                                         MROWS, EMB, EMB);
        cvtT(Wv + wEE, wbuf, EMB, EMB, stream);
        gemm_kernel<GF_OUTBF><<<gemmE, 256, 0, stream>>>(h, wbuf, nullptr, nullptr, v,
                                                         MROWS, EMB, EMB);

        attn_kernel<<<dim3(SEQ / 64, NHEAD, BATCH), 128, 0, stream>>>(q, k, v, attn,
                                                                      scale);

        cvtT(Wo + wEE, wbuf, EMB, EMB, stream);
        gemm_kernel<GF_RES><<<gemmE, 256, 0, stream>>>(attn, wbuf, nullptr, x, x,
                                                       MROWS, EMB, EMB);

        ln_kernel<<<MROWS, 256, 0, stream>>>(x, ln2_g + l * EMB, ln2_b + l * EMB, h);

        cvtT(W1 + (size_t)l * EMB * FFD, wbuf, FFD, EMB, stream);
        gemm_kernel<GF_BIAS | GF_RELU | GF_OUTBF><<<gemmFF, 256, 0, stream>>>(
            h, wbuf, b1 + (size_t)l * FFD, nullptr, ff, MROWS, FFD, EMB);

        cvtT(W2 + (size_t)l * FFD * EMB, wbuf, EMB, FFD, stream);
        gemm_kernel<GF_BIAS | GF_RES><<<gemmE, 256, 0, stream>>>(
            ff, wbuf, b2 + (size_t)l * EMB, x, x, MROWS, EMB, FFD);
    }

    // logits: convert final residual stream to bf16, then big GEMM with bias
    cvt(x, h, NTE, stream);
    cvtT(lmh_w, wbuf, VOCAB, EMB, stream);
    gemm_kernel<GF_BIAS><<<gemmV, 256, 0, stream>>>(h, wbuf, lmh_b, nullptr, out,
                                                    MROWS, VOCAB, EMB);
}
